// ModifiedMSA_8358006358108
// MI455X (gfx1250) — compile-verified, matrix operand order checked by reading, not run on hardware
//
#include <hip/hip_runtime.h>
#include <hip/hip_bf16.h>

// ---------------------------------------------------------------------------
// Nystromformer MSA for MI455X (gfx1250, wave32, WMMA bf16 16x16x32).
// out = [softmax(q kt^T)] @ (softmax(qt kt^T)^-1 @ (softmax(qt k^T) @ v)) @ Wo + bo
// Heavy GEMMs -> v_wmma_f32_16x16x32_bf16 (f32 accum); contiguous tile loads
// -> global_load_async_to_lds_b128 (ASYNCcnt path), transposes staged via VGPR.
// ---------------------------------------------------------------------------

#define BB 8
#define NN 4096
#define EE 768
#define HH 12
#define MM 64
#define DD 64
#define BH (BB * HH)   // 96
#define BN (BB * NN)   // 32768
#define E3 (3 * EE)    // 2304

typedef __bf16 bf16_t;
typedef __attribute__((ext_vector_type(16))) __bf16 v16bf;
typedef __attribute__((ext_vector_type(8)))  __bf16 v8bf;
typedef __attribute__((ext_vector_type(4)))  __bf16 v4bf;
typedef __attribute__((ext_vector_type(8)))  float  v8f;

#define LDT 72  // padded LDS row stride (bf16 elems); keeps 16B alignment for b128

#define AS(n) __attribute__((address_space(n)))
#define USE_ASYNC 1

// 16B global -> LDS copy. Async path uses the CDNA5 async-DMA instruction
// (tracked by ASYNCcnt, no VGPR round trip); must be followed by
// cp_async_join() before the workgroup barrier.
__device__ __forceinline__ void cp_async16(bf16_t* l, const bf16_t* g) {
#if USE_ASYNC
  unsigned laddr = (unsigned)(unsigned long long)(AS(3) bf16_t*)l;  // 32-bit LDS addr
  unsigned long long gaddr = (unsigned long long)g;                 // flat==global
  asm volatile("global_load_async_to_lds_b128 %0, %1, off"
               :: "v"(laddr), "v"(gaddr)
               : "memory");
#else
  *(v8bf*)l = *(const v8bf*)g;
#endif
}
__device__ __forceinline__ void cp_async_join() {
#if USE_ASYNC
  asm volatile("s_wait_asynccnt 0x0" ::: "memory");
#endif
}

// Load one 16x32 bf16 WMMA operand fragment from an LDS tile laid out [row][k].
__device__ __forceinline__ v16bf ld_frag(const bf16_t* lds, int rowBase, int k0) {
  int lane = threadIdx.x & 31;
  int r = lane & 15;
  int g = lane >> 4;
  const bf16_t* p = lds + (rowBase + r) * LDT + k0 + g * 8;
  v8bf lo = *(const v8bf*)p;
  v8bf hi = *(const v8bf*)(p + 16);
  v16bf o;
#pragma unroll
  for (int i = 0; i < 8; ++i) { o[i] = lo[i]; o[i + 8] = hi[i]; }
  return o;
}

__device__ __forceinline__ v8f wmma_bf16(v16bf a, v16bf b, v8f c) {
  return __builtin_amdgcn_wmma_f32_16x16x32_bf16(false, a, false, b, (short)0, c,
                                                 false, false);
}

// Hoist all fragment loads, then run the 8-WMMA chain (4 N-tiles x K=64).
__device__ __forceinline__ void mma_64x64_k64(const bf16_t* As, const bf16_t* Bt,
                                              int wv, v8f acc[4]) {
  v16bf a0 = ld_frag(As, wv * 16, 0);
  v16bf a1 = ld_frag(As, wv * 16, 32);
  v16bf b0[4], b1[4];
#pragma unroll
  for (int nt = 0; nt < 4; ++nt) {
    b0[nt] = ld_frag(Bt, nt * 16, 0);
    b1[nt] = ld_frag(Bt, nt * 16, 32);
  }
#pragma unroll
  for (int nt = 0; nt < 4; ++nt) {
    acc[nt] = wmma_bf16(a0, b0[nt], acc[nt]);
    acc[nt] = wmma_bf16(a1, b1[nt], acc[nt]);
  }
}

// --------------------------- f32 -> bf16 convert ---------------------------
__global__ void cvt_bf16(const float* __restrict__ in, bf16_t* __restrict__ out, int n) {
  int i = blockIdx.x * blockDim.x + threadIdx.x;
  int stride = gridDim.x * blockDim.x;
  for (; i < n; i += stride) out[i] = (bf16_t)in[i];
}

// --------------------------- QKV projection GEMM ---------------------------
__global__ __launch_bounds__(128) void gemm_qkv(const bf16_t* __restrict__ X,
                                                const bf16_t* __restrict__ W,
                                                bf16_t* __restrict__ Q,
                                                bf16_t* __restrict__ K,
                                                bf16_t* __restrict__ V) {
  __shared__ __align__(16) bf16_t As[64 * LDT];
  __shared__ __align__(16) bf16_t Bt[64 * LDT];
  const int colBase = blockIdx.x * 64;
  const int rowBase = blockIdx.y * 64;
  const int t = threadIdx.x, wv = t >> 5, lane = t & 31;
  v8f acc[4] = {};
  for (int k0 = 0; k0 < EE; k0 += 64) {
#pragma unroll
    for (int i = 0; i < 4; ++i) {           // A tile 64x64: async DMA to LDS
      int c = t + i * 128;
      int row = c >> 3, kg = (c & 7) * 8;
      cp_async16(&As[row * LDT + kg], &X[(rowBase + row) * EE + k0 + kg]);
    }
#pragma unroll
    for (int i = 0; i < 8; ++i) {           // B tile transposed: Bt[col][k]
      int c = t + i * 128;
      int k = c >> 4, cg = (c & 15) * 4;
      v4bf w4 = *(const v4bf*)&W[(k0 + k) * E3 + colBase + cg];
#pragma unroll
      for (int j = 0; j < 4; ++j) Bt[(cg + j) * LDT + k] = w4[j];
    }
    if (k0 + 64 < EE)
      __builtin_prefetch(&W[(k0 + 64 + (t >> 2)) * E3 + colBase], 0, 1);
    cp_async_join();
    __syncthreads();
    mma_64x64_k64(As, Bt, wv, acc);
    __syncthreads();
  }
  const int which = colBase / EE;           // 0=q 1=k 2=v
  const int h = (colBase % EE) >> 6;
  bf16_t* dst = (which == 0) ? Q : (which == 1) ? K : V;
  const float scale = (which == 0) ? 0.125f : 1.0f;
#pragma unroll
  for (int nt = 0; nt < 4; ++nt) {
    int d = nt * 16 + (lane & 15);
#pragma unroll
    for (int i = 0; i < 8; ++i) {
      int rm = wv * 16 + ((lane >> 4) ? 8 : 0) + i;
      int n = rowBase + rm;
      int b = n >> 12, nn = n & (NN - 1);
      dst[((size_t)(b * HH + h) * NN + nn) * DD + d] = (bf16_t)(acc[nt][i] * scale);
    }
  }
}

// --------------------------- landmark means --------------------------------
__global__ void landmarks(const bf16_t* __restrict__ Q, const bf16_t* __restrict__ K,
                          bf16_t* __restrict__ QT, bf16_t* __restrict__ KT,
                          float* __restrict__ QTf, float* __restrict__ KTf) {
  int idx = blockIdx.x * blockDim.x + threadIdx.x;
  if (idx >= BH * MM * DD) return;
  int d = idx & 63;
  int m = (idx >> 6) & 63;
  int bh = idx >> 12;
  const bf16_t* qp = Q + ((size_t)bh * NN + m * 64) * DD + d;
  const bf16_t* kp = K + ((size_t)bh * NN + m * 64) * DD + d;
  float sq = 0.f, sk = 0.f;
  for (int s = 0; s < 64; ++s) { sq += (float)qp[s * DD]; sk += (float)kp[s * DD]; }
  sq *= (1.f / 64.f); sk *= (1.f / 64.f);
  QT[idx] = (bf16_t)sq; KT[idx] = (bf16_t)sk;
  QTf[idx] = sq;        KTf[idx] = sk;
}

// ----------------- kernel2 = softmax(qt kt^T); invert (Gauss-Jordan) -------
__global__ __launch_bounds__(256) void kernel2_invert(const float* __restrict__ QTf,
                                                      const float* __restrict__ KTf,
                                                      bf16_t* __restrict__ K2INV) {
  __shared__ float Aug[64][130];
  __shared__ float fcol[64];
  const int bh = blockIdx.x, t = threadIdx.x;
  const float* qt = QTf + (size_t)bh * MM * DD;
  const float* kt = KTf + (size_t)bh * MM * DD;
  for (int e = t; e < 64 * 64; e += 256) {
    int r = e >> 6, c = e & 63;
    float s = 0.f;
    for (int d = 0; d < 64; ++d) s += qt[r * 64 + d] * kt[c * 64 + d];
    Aug[r][c] = s;
  }
  __syncthreads();
  if (t < 64) {                              // row softmax
    float mx = -1e30f;
    for (int c = 0; c < 64; ++c) mx = fmaxf(mx, Aug[t][c]);
    float sum = 0.f;
    for (int c = 0; c < 64; ++c) { float e = __expf(Aug[t][c] - mx); Aug[t][c] = e; sum += e; }
    float inv = 1.f / sum;
    for (int c = 0; c < 64; ++c) Aug[t][c] *= inv;
  }
  __syncthreads();
  for (int e = t; e < 64 * 64; e += 256) {   // augment identity
    int r = e >> 6, c = e & 63;
    Aug[r][64 + c] = (r == c) ? 1.f : 0.f;
  }
  __syncthreads();
  for (int p = 0; p < 64; ++p) {             // no pivoting (softmax matrix)
    float piv = Aug[p][p];
    __syncthreads();
    float inv = 1.f / piv;
    if (t < 128) Aug[p][t] *= inv;
    if (t < 64) fcol[t] = (t == p) ? 0.f : Aug[t][p];
    __syncthreads();
    for (int e = t; e < 64 * 128; e += 256) {
      int r = e >> 7, c = e & 127;
      if (r != p) Aug[r][c] -= fcol[r] * Aug[p][c];
    }
    __syncthreads();
  }
  for (int e = t; e < 64 * 64; e += 256) {
    int r = e >> 6, c = e & 63;
    K2INV[(size_t)bh * MM * MM + r * 64 + c] = (bf16_t)Aug[r][64 + c];
  }
}

// ----------------- kernel1 = softmax(q kt^T) over M=64 ---------------------
__global__ __launch_bounds__(128) void kernel1_softmax(const bf16_t* __restrict__ Q,
                                                       const bf16_t* __restrict__ KT,
                                                       bf16_t* __restrict__ P1) {
  __shared__ __align__(16) bf16_t As[64 * LDT];
  __shared__ __align__(16) bf16_t Bt[64 * LDT];
  __shared__ float Cs[64][66];
  const int bh = blockIdx.x, nb = blockIdx.y;
  const int t = threadIdx.x, wv = t >> 5, lane = t & 31;
  const bf16_t* q  = Q  + ((size_t)bh * NN + nb * 64) * DD;
  const bf16_t* kt = KT + (size_t)bh * MM * DD;
#pragma unroll
  for (int i = 0; i < 4; ++i) {
    int c = t + i * 128;
    int row = c >> 3, kg = (c & 7) * 8;
    cp_async16(&As[row * LDT + kg], &q[row * DD + kg]);
    // logits = q @ kt^T: transposed B tile is kt rows verbatim -> async too.
    cp_async16(&Bt[row * LDT + kg], &kt[row * DD + kg]);
  }
  cp_async_join();
  __syncthreads();
  v8f acc[4] = {};
  mma_64x64_k64(As, Bt, wv, acc);
#pragma unroll
  for (int nt = 0; nt < 4; ++nt)
#pragma unroll
    for (int i = 0; i < 8; ++i) {
      int rm = wv * 16 + ((lane >> 4) ? 8 : 0) + i;
      Cs[rm][nt * 16 + (lane & 15)] = acc[nt][i];
    }
  __syncthreads();
  if (t < 64) {
    float mx = -1e30f;
    for (int c = 0; c < 64; ++c) mx = fmaxf(mx, Cs[t][c]);
    float sum = 0.f;
    for (int c = 0; c < 64; ++c) { float e = __expf(Cs[t][c] - mx); Cs[t][c] = e; sum += e; }
    float inv = 1.f / sum;
    bf16_t* out = P1 + ((size_t)bh * NN + nb * 64 + t) * MM;
    for (int c = 0; c < 64; ++c) out[c] = (bf16_t)(Cs[t][c] * inv);
  }
}

// ----------------- kernel3 = softmax(qt k^T) over N=4096 (fused 2-pass) ----
__global__ __launch_bounds__(128) void kernel3_softmax(const bf16_t* __restrict__ QT,
                                                       const bf16_t* __restrict__ Kb,
                                                       bf16_t* __restrict__ P3) {
  __shared__ __align__(16) bf16_t As[64 * LDT];
  __shared__ __align__(16) bf16_t Bt[64 * LDT];
  __shared__ float Cs[64][66];
  __shared__ float rowM[64], rowS[64];
  const int bh = blockIdx.x;
  const int t = threadIdx.x, wv = t >> 5, lane = t & 31;
  const bf16_t* qt = QT + (size_t)bh * MM * DD;
  const bf16_t* kg = Kb + (size_t)bh * NN * DD;
#pragma unroll
  for (int i = 0; i < 4; ++i) {
    int c = t + i * 128;
    int row = c >> 3, kk = (c & 7) * 8;
    cp_async16(&As[row * LDT + kk], &qt[row * DD + kk]);
  }
  if (t < 64) { rowM[t] = -1e30f; rowS[t] = 0.f; }
  cp_async_join();
  __syncthreads();
  v16bf a0 = ld_frag(As, wv * 16, 0);
  v16bf a1 = ld_frag(As, wv * 16, 32);
  for (int pass = 0; pass < 2; ++pass) {
    for (int nb = 0; nb < NN / 64; ++nb) {
#pragma unroll
      for (int i = 0; i < 4; ++i) {          // Bt[col=n][k=d] = k rows verbatim
        int c = t + i * 128;
        int row = c >> 3, kk = (c & 7) * 8;
        cp_async16(&Bt[row * LDT + kk], &kg[(nb * 64 + row) * DD + kk]);
      }
      cp_async_join();
      __syncthreads();
      v8f acc[4] = {};
      v16bf b0[4], b1[4];
#pragma unroll
      for (int nt = 0; nt < 4; ++nt) {
        b0[nt] = ld_frag(Bt, nt * 16, 0);
        b1[nt] = ld_frag(Bt, nt * 16, 32);
      }
#pragma unroll
      for (int nt = 0; nt < 4; ++nt) {
        acc[nt] = wmma_bf16(a0, b0[nt], acc[nt]);
        acc[nt] = wmma_bf16(a1, b1[nt], acc[nt]);
      }
#pragma unroll
      for (int nt = 0; nt < 4; ++nt)
#pragma unroll
        for (int i = 0; i < 8; ++i) {
          int rm = wv * 16 + ((lane >> 4) ? 8 : 0) + i;
          Cs[rm][nt * 16 + (lane & 15)] = acc[nt][i];
        }
      __syncthreads();
      if (pass == 0) {
        if (t < 64) {                        // online max/sum update
          float m = rowM[t], s = rowS[t];
          float cm = -1e30f;
          for (int c = 0; c < 64; ++c) cm = fmaxf(cm, Cs[t][c]);
          float mn = fmaxf(m, cm);
          float cs = 0.f;
          for (int c = 0; c < 64; ++c) cs += __expf(Cs[t][c] - mn);
          rowS[t] = s * __expf(m - mn) + cs;
          rowM[t] = mn;
        }
      } else {
        if (t < 64) {
          float m = rowM[t], inv = 1.f / rowS[t];
          bf16_t* out = P3 + ((size_t)bh * MM + t) * NN + nb * 64;
          for (int c = 0; c < 64; ++c) out[c] = (bf16_t)(__expf(Cs[t][c] - m) * inv);
        }
      }
      __syncthreads();
    }
  }
}

// ----------------- S3V = kernel3 @ v  (64 x 64, K = 4096) ------------------
__global__ __launch_bounds__(128) void gemm_s3v(const bf16_t* __restrict__ P3,
                                                const bf16_t* __restrict__ V,
                                                float* __restrict__ S3V) {
  __shared__ __align__(16) bf16_t As[64 * LDT];
  __shared__ __align__(16) bf16_t Bt[64 * LDT];
  const int bh = blockIdx.x;
  const int t = threadIdx.x, wv = t >> 5, lane = t & 31;
  const bf16_t* p3 = P3 + (size_t)bh * MM * NN;
  const bf16_t* v  = V  + (size_t)bh * NN * DD;
  v8f acc[4] = {};
  for (int k0 = 0; k0 < NN; k0 += 64) {
#pragma unroll
    for (int i = 0; i < 4; ++i) {
      int c = t + i * 128;
      int row = c >> 3, kk = (c & 7) * 8;
      cp_async16(&As[row * LDT + kk], &p3[row * NN + k0 + kk]);
    }
#pragma unroll
    for (int i = 0; i < 8; ++i) {            // Bt[col=d][k=n] = v[n][d] transpose
      int c = t + i * 128;
      int k = c >> 4, cg = (c & 15) * 4;
      v4bf w4 = *(const v4bf*)&v[(k0 + k) * DD + cg];
#pragma unroll
      for (int j = 0; j < 4; ++j) Bt[(cg + j) * LDT + k] = w4[j];
    }
    cp_async_join();
    __syncthreads();
    mma_64x64_k64(As, Bt, wv, acc);
    __syncthreads();
  }
  float* out = S3V + (size_t)bh * MM * DD;
#pragma unroll
  for (int nt = 0; nt < 4; ++nt)
#pragma unroll
    for (int i = 0; i < 8; ++i) {
      int rm = wv * 16 + ((lane >> 4) ? 8 : 0) + i;
      out[rm * DD + nt * 16 + (lane & 15)] = acc[nt][i];
    }
}

// ----------------- W = kernel2^-1 @ S3V  (64x64x64) ------------------------
__global__ __launch_bounds__(128) void gemm_w(const bf16_t* __restrict__ K2INV,
                                              const float* __restrict__ S3V,
                                              bf16_t* __restrict__ Wm) {
  __shared__ __align__(16) bf16_t As[64 * LDT];
  __shared__ __align__(16) bf16_t Bt[64 * LDT];
  const int bh = blockIdx.x;
  const int t = threadIdx.x, wv = t >> 5, lane = t & 31;
  const bf16_t* a = K2INV + (size_t)bh * MM * MM;
  const float*  s = S3V   + (size_t)bh * MM * DD;
#pragma unroll
  for (int i = 0; i < 4; ++i) {
    int c = t + i * 128;
    int row = c >> 3, kk = (c & 7) * 8;
    cp_async16(&As[row * LDT + kk], &a[row * MM + kk]);
  }
  for (int e = t; e < 64 * 64; e += 128) {   // Bt[col=d][k=m] = S3V[m][d]
    int m = e >> 6, d = e & 63;
    Bt[d * LDT + m] = (bf16_t)s[m * DD + d];
  }
  cp_async_join();
  __syncthreads();
  v8f acc[4] = {};
  mma_64x64_k64(As, Bt, wv, acc);
  bf16_t* out = Wm + (size_t)bh * MM * DD;
#pragma unroll
  for (int nt = 0; nt < 4; ++nt)
#pragma unroll
    for (int i = 0; i < 8; ++i) {
      int rm = wv * 16 + ((lane >> 4) ? 8 : 0) + i;
      out[rm * DD + nt * 16 + (lane & 15)] = (bf16_t)acc[nt][i];
    }
}

// ----------------- attn_out = kernel1 @ W -> (B,N,E) bf16 ------------------
__global__ __launch_bounds__(128) void gemm_attnout(const bf16_t* __restrict__ P1,
                                                    const bf16_t* __restrict__ Wm,
                                                    bf16_t* __restrict__ AO) {
  __shared__ __align__(16) bf16_t As[64 * LDT];
  __shared__ __align__(16) bf16_t Bt[64 * LDT];
  const int bh = blockIdx.x, nb = blockIdx.y;
  const int b = bh / HH, h = bh % HH;
  const int t = threadIdx.x, wv = t >> 5, lane = t & 31;
  const bf16_t* p1 = P1 + ((size_t)bh * NN + nb * 64) * MM;
  const bf16_t* w  = Wm + (size_t)bh * MM * DD;
#pragma unroll
  for (int i = 0; i < 4; ++i) {
    int c = t + i * 128;
    int row = c >> 3, kk = (c & 7) * 8;
    cp_async16(&As[row * LDT + kk], &p1[row * MM + kk]);
  }
#pragma unroll
  for (int i = 0; i < 8; ++i) {              // Bt[col=d][k=m] = W[m][d]
    int c = t + i * 128;
    int m = c >> 4, dg = (c & 15) * 4;
    v4bf w4 = *(const v4bf*)&w[m * DD + dg];
#pragma unroll
    for (int j = 0; j < 4; ++j) Bt[(dg + j) * LDT + m] = w4[j];
  }
  cp_async_join();
  __syncthreads();
  v8f acc[4] = {};
  mma_64x64_k64(As, Bt, wv, acc);
#pragma unroll
  for (int nt = 0; nt < 4; ++nt)
#pragma unroll
    for (int i = 0; i < 8; ++i) {
      int rm = wv * 16 + ((lane >> 4) ? 8 : 0) + i;
      int n = nb * 64 + rm;
      int d = nt * 16 + (lane & 15);
      AO[((size_t)b * NN + n) * EE + h * DD + d] = (bf16_t)acc[nt][i];
    }
}

// ----------------- final projection: out = attn @ Wo + bo (f32 out) --------
__global__ __launch_bounds__(128) void gemm_out(const bf16_t* __restrict__ A,
                                                const bf16_t* __restrict__ W,
                                                const float* __restrict__ bias,
                                                float* __restrict__ Out) {
  __shared__ __align__(16) bf16_t As[64 * LDT];
  __shared__ __align__(16) bf16_t Bt[64 * LDT];
  const int colBase = blockIdx.x * 64;
  const int rowBase = blockIdx.y * 64;
  const int t = threadIdx.x, wv = t >> 5, lane = t & 31;
  v8f acc[4] = {};
  for (int k0 = 0; k0 < EE; k0 += 64) {
#pragma unroll
    for (int i = 0; i < 4; ++i) {
      int c = t + i * 128;
      int row = c >> 3, kg = (c & 7) * 8;
      cp_async16(&As[row * LDT + kg], &A[(size_t)(rowBase + row) * EE + k0 + kg]);
    }
#pragma unroll
    for (int i = 0; i < 8; ++i) {
      int c = t + i * 128;
      int k = c >> 4, cg = (c & 15) * 4;
      v4bf w4 = *(const v4bf*)&W[(size_t)(k0 + k) * EE + colBase + cg];
#pragma unroll
      for (int j = 0; j < 4; ++j) Bt[(cg + j) * LDT + k] = w4[j];
    }
    if (k0 + 64 < EE)
      __builtin_prefetch(&W[(size_t)(k0 + 64 + (t >> 2)) * EE + colBase], 0, 1);
    cp_async_join();
    __syncthreads();
    mma_64x64_k64(As, Bt, wv, acc);
    __syncthreads();
  }
#pragma unroll
  for (int nt = 0; nt < 4; ++nt) {
    int col = colBase + nt * 16 + (lane & 15);
    float bcol = bias[col];
#pragma unroll
    for (int i = 0; i < 8; ++i) {
      int rm = wv * 16 + ((lane >> 4) ? 8 : 0) + i;
      Out[(size_t)(rowBase + rm) * EE + col] = acc[nt][i] + bcol;
    }
  }
}

// ---------------------------------------------------------------------------
extern "C" void kernel_launch(void* const* d_in, const int* in_sizes, int n_in,
                              void* d_out, int out_size, void* d_ws, size_t ws_size,
                              hipStream_t stream) {
  (void)in_sizes; (void)n_in; (void)out_size;
  const float* x    = (const float*)d_in[0];
  const float* Wqkv = (const float*)d_in[1];
  const float* Wo   = (const float*)d_in[2];
  const float* bo   = (const float*)d_in[3];
  float* out = (float*)d_out;

  char* ws = (char*)d_ws;
  size_t off = 0;
  auto alloc = [&](size_t bytes) -> void* {
    void* p = ws + off;
    off = (off + bytes + 255) & ~(size_t)255;
    return p;
  };
  bf16_t* xattn_bf = (bf16_t*)alloc((size_t)BN * EE * 2);   // x_bf; later attn out
  bf16_t* wqkv_bf  = (bf16_t*)alloc((size_t)EE * E3 * 2);
  bf16_t* wo_bf    = (bf16_t*)alloc((size_t)EE * EE * 2);
  bf16_t* q_bf     = (bf16_t*)alloc((size_t)BH * NN * DD * 2);
  bf16_t* k_bf     = (bf16_t*)alloc((size_t)BH * NN * DD * 2);
  bf16_t* v_bf     = (bf16_t*)alloc((size_t)BH * NN * DD * 2);
  bf16_t* qt_bf    = (bf16_t*)alloc((size_t)BH * MM * DD * 2);
  bf16_t* kt_bf    = (bf16_t*)alloc((size_t)BH * MM * DD * 2);
  float*  qt_f     = (float*) alloc((size_t)BH * MM * DD * 4);
  float*  kt_f     = (float*) alloc((size_t)BH * MM * DD * 4);
  bf16_t* k2inv_bf = (bf16_t*)alloc((size_t)BH * MM * MM * 2);
  bf16_t* p1_bf    = (bf16_t*)alloc((size_t)BH * NN * MM * 2);
  bf16_t* p3_bf    = (bf16_t*)alloc((size_t)BH * MM * NN * 2);
  float*  s3v_f    = (float*) alloc((size_t)BH * MM * DD * 4);
  bf16_t* w_bf     = (bf16_t*)alloc((size_t)BH * MM * DD * 2);
  if (off > ws_size) return;  // workspace too small; bail deterministically

  cvt_bf16<<<4096, 256, 0, stream>>>(x,    xattn_bf, BN * EE);
  cvt_bf16<<<4096, 256, 0, stream>>>(Wqkv, wqkv_bf,  EE * E3);
  cvt_bf16<<<2048, 256, 0, stream>>>(Wo,   wo_bf,    EE * EE);

  dim3 gQKV(E3 / 64, BN / 64);                       // 36 x 512
  gemm_qkv<<<gQKV, 128, 0, stream>>>(xattn_bf, wqkv_bf, q_bf, k_bf, v_bf);

  landmarks<<<(BH * MM * DD + 255) / 256, 256, 0, stream>>>(q_bf, k_bf, qt_bf, kt_bf,
                                                            qt_f, kt_f);
  kernel2_invert<<<BH, 256, 0, stream>>>(qt_f, kt_f, k2inv_bf);

  dim3 gP1(BH, NN / 64);                             // 96 x 64
  kernel1_softmax<<<gP1, 128, 0, stream>>>(q_bf, kt_bf, p1_bf);
  kernel3_softmax<<<BH, 128, 0, stream>>>(qt_bf, k_bf, p3_bf);

  gemm_s3v<<<BH, 128, 0, stream>>>(p3_bf, v_bf, s3v_f);
  gemm_w<<<BH, 128, 0, stream>>>(k2inv_bf, s3v_f, w_bf);

  dim3 gAO(BH, NN / 64);
  gemm_attnout<<<gAO, 128, 0, stream>>>(p1_bf, w_bf, xattn_bf);

  dim3 gO(EE / 64, BN / 64);                         // 12 x 512
  gemm_out<<<gO, 128, 0, stream>>>(xattn_bf, wo_bf, bo, out);
}
